// FullCovPredictor_63101659513388
// MI455X (gfx1250) — matrix operand order repaired and checked by reading, not compiled, not run on hardware
//
#include <hip/hip_runtime.h>
#include <math.h>

typedef __attribute__((ext_vector_type(16))) __bf16 v16bf;
typedef __attribute__((ext_vector_type(4)))  __bf16 v4bf;
typedef __attribute__((ext_vector_type(8)))  float  v8f;
typedef __attribute__((ext_vector_type(4)))  unsigned int u32x4;
typedef __attribute__((ext_vector_type(4)))  float  f32x4;

union Frag16 { v16bf v; u32x4 q[2]; };

#define TOK_PER_BLK 64
#define XA_STRIDE   264   // 256 + 8 bf16 pad -> conflict-free A-fragment ds_load_b128
#define H1_STRIDE   1032  // 1024 + 8 bf16 pad
#define HB_STRIDE   260   // 256 + 4 f32 pad (keeps float4 alignment)
#define COV_EPS     1e-8f

// ---------------------------------------------------------------------------
// Repack f32 row-major weight (K x N) into bf16 WMMA B-fragment layout.
// Fragment (nt, kt) covers B[kt*32 .. kt*32+31][nt*16 .. nt*16+15], stored as
// 512 contiguous bf16: element index = lane*16 + e, where
//   col = lane & 15 ; kbase = (lane >= 16) ? 8 : 0 ; K = kt*32 + kbase + e + (e >= 8 ? 8 : 0)
// (matches ISA 16-bit 32x16 B layout -> two global_load_b128 per fragment)
// ---------------------------------------------------------------------------
__global__ void repack_w_bf16(const float* __restrict__ W, __bf16* __restrict__ out,
                              int K, int N) {
  int idx = blockIdx.x * blockDim.x + threadIdx.x;
  int total = K * N;
  if (idx >= total) return;
  int frag   = idx >> 9;
  int within = idx & 511;
  int lane   = within >> 4;
  int e      = within & 15;
  int ktiles = K >> 5;
  int nt = frag / ktiles;
  int kt = frag - nt * ktiles;
  int col   = lane & 15;
  int kbase = (lane >> 4) << 3;
  int k = kt * 32 + kbase + e + ((e >> 3) << 3);
  int n = nt * 16 + col;
  out[idx] = (__bf16)W[(size_t)k * N + n];
}

// Branch-free tanh-GELU: 0.5*v*(1+tanh(u)) == v * sigmoid(2u).
// Tails: u->+inf => exp(-2u)->0 => v ; u->-inf => rcp(inf)=0 => 0. No NaN path.
__device__ __forceinline__ float gelu_fast(float v) {
  float u2 = 1.5957691216057308f * (v + 0.044715f * v * v * v);
  return v * __builtin_amdgcn_rcpf(1.0f + __expf(-u2));
}

__device__ __forceinline__ float softplus_f(float v) {
  return (v > 20.0f) ? v : __logf(1.0f + __expf(v));
}

// ---------------------------------------------------------------------------
// Fused: h = x + gelu(x@W1+b1)@W2 + b2 ; heads ; 3x3 MGS-QR ; cov = Q L Q^T.
// 64 tokens per block, 8 wave32. Each wave owns a fixed 16-row m-tile.
// ---------------------------------------------------------------------------
__global__ __launch_bounds__(256) void fused_cov_kernel(
    const float* __restrict__ x,
    const float* __restrict__ b1,
    const float* __restrict__ b2,
    const float* __restrict__ bval,
    const float* __restrict__ bvec,
    const float* __restrict__ Wval,
    const float* __restrict__ Wvec,
    const __bf16* __restrict__ w1p,
    const __bf16* __restrict__ w2p,
    float* __restrict__ out)
{
  extern __shared__ char smem[];
  __bf16* xa   = (__bf16*)smem;                          // 64 x 264  bf16
  __bf16* h1   = xa + TOK_PER_BLK * XA_STRIDE;           // 64 x 1032 bf16
  float*  xf   = (float*)(h1 + TOK_PER_BLK * H1_STRIDE); // 64 x 260  f32
  float*  hb   = xf + TOK_PER_BLK * HB_STRIDE;           // 64 x 260  f32
  float*  wh   = hb + TOK_PER_BLK * HB_STRIDE;           // 256 x 12  f32
  float*  vout = wh + 256 * 12;                          // 64 x 13   f32

  const int tid  = threadIdx.x;
  const int lane = tid & 31;
  const int wave = tid >> 5;
  const int tokBase = blockIdx.x * TOK_PER_BLK;

  // ---- Phase 0: stage x (f32 + bf16) and head weights into LDS -----------
  for (int i = tid; i < TOK_PER_BLK * 64; i += 256) {   // 64 float4 per token row
    int m  = i >> 6;
    int k4 = (i & 63) << 2;
    f32x4 v = *(const f32x4*)(x + (size_t)(tokBase + m) * 256 + k4);
    *(f32x4*)(xf + m * HB_STRIDE + k4) = v;
    v4bf bv = { (__bf16)v[0], (__bf16)v[1], (__bf16)v[2], (__bf16)v[3] };
    *(v4bf*)(xa + m * XA_STRIDE + k4) = bv;             // ds_store_b64, 8B aligned
  }
  for (int i = tid; i < 256 * 3; i += 256)
    wh[(i / 3) * 12 + (i % 3)] = Wval[i];
  for (int i = tid; i < 256 * 9; i += 256)
    wh[(i / 9) * 12 + 3 + (i % 9)] = Wvec[i];
  __syncthreads();

  const int arow  = lane & 15;
  const int aklo  = (lane >> 4) << 3;  // A-fragment K sub-offset (0 or 8)
  const int ccol  = lane & 15;         // C/D layout: column per lane
  const int cmoff = (lane >> 4) << 3;  // +0 (lanes 0-15) / +8 (lanes 16-31)

  // ---- Phase 1: H1 = gelu(x @ W1 + b1) -----------------------------------
  // wave -> fixed mt = wave&3, nt range of 32. A fragments loaded ONCE.
  {
    const int mt     = wave & 3;
    const int ntbase = (wave >> 2) * 32;
    Frag16 a[8];
#pragma unroll
    for (int kt = 0; kt < 8; ++kt) {
      int aoff = (mt * 16 + arow) * XA_STRIDE + kt * 32 + aklo;
      a[kt].q[0] = *(const u32x4*)(xa + aoff);
      a[kt].q[1] = *(const u32x4*)(xa + aoff + 16);
    }
    for (int nti = 0; nti < 32; ++nti) {
      int nt = ntbase + nti;
      v8f c = {};
#pragma unroll
      for (int kt = 0; kt < 8; ++kt) {
        Frag16 bm;
        const u32x4* bp = (const u32x4*)(w1p + ((size_t)(nt * 8 + kt) << 9) + lane * 16);
        bm.q[0] = bp[0];
        bm.q[1] = bp[1];
        c = __builtin_amdgcn_wmma_f32_16x16x32_bf16(false, a[kt].v, false, bm.v,
                                                    (short)0, c, false, false);
      }
      int coln  = nt * 16 + ccol;
      float bias = b1[coln];
      int mbase = mt * 16 + cmoff;
#pragma unroll
      for (int r = 0; r < 8; ++r)
        h1[(mbase + r) * H1_STRIDE + coln] = (__bf16)gelu_fast(c[r] + bias);
    }
  }
  __syncthreads();

  // ---- Phase 2: h = x + H1 @ W2 + b2 -------------------------------------
  // wave -> fixed mt, 8 n-tiles held as live C accumulators; K streamed in
  // blocks of 8 fragments (A loaded once per block, reused by 8 n-tiles).
  {
    const int mt     = wave & 3;
    const int ntbase = (wave >> 2) * 8;
    v8f c[8] = {};
    for (int ktb = 0; ktb < 4; ++ktb) {
      Frag16 a[8];
#pragma unroll
      for (int kk = 0; kk < 8; ++kk) {
        int kt   = ktb * 8 + kk;
        int aoff = (mt * 16 + arow) * H1_STRIDE + kt * 32 + aklo;
        a[kk].q[0] = *(const u32x4*)(h1 + aoff);
        a[kk].q[1] = *(const u32x4*)(h1 + aoff + 16);
      }
#pragma unroll
      for (int j = 0; j < 8; ++j) {
        int nt = ntbase + j;
#pragma unroll
        for (int kk = 0; kk < 8; ++kk) {
          int kt = ktb * 8 + kk;
          Frag16 bm;
          const u32x4* bp = (const u32x4*)(w2p + ((size_t)(nt * 32 + kt) << 9) + lane * 16);
          bm.q[0] = bp[0];
          bm.q[1] = bp[1];
          c[j] = __builtin_amdgcn_wmma_f32_16x16x32_bf16(false, a[kk].v, false, bm.v,
                                                         (short)0, c[j], false, false);
        }
      }
    }
    int mbase = mt * 16 + cmoff;
#pragma unroll
    for (int j = 0; j < 8; ++j) {
      int coln  = (ntbase + j) * 16 + ccol;
      float bias = b2[coln];
#pragma unroll
      for (int r = 0; r < 8; ++r) {
        int m = mbase + r;
        hb[m * HB_STRIDE + coln] = c[j][r] + bias + xf[m * HB_STRIDE + coln];
      }
    }
  }
  __syncthreads();

  // ---- Phase 3: heads. 4 threads per token, 3 dot products each ----------
  {
    int t = tid >> 2;
    int g = tid & 3;
    int o = g * 3;
    float a0 = 0.f, a1 = 0.f, a2 = 0.f;
    const float* hr = hb + t * HB_STRIDE;
    for (int k = 0; k < 256; k += 4) {
      f32x4 hv = *(const f32x4*)(hr + k);
#pragma unroll
      for (int i = 0; i < 4; ++i) {
        float hk = hv[i];
        const float* w = wh + (k + i) * 12 + o;
        a0 = fmaf(hk, w[0], a0);
        a1 = fmaf(hk, w[1], a1);
        a2 = fmaf(hk, w[2], a2);
      }
    }
    float r0, r1, r2;
    if (g == 0) {
      r0 = softplus_f(a0 + bval[0]);
      r1 = softplus_f(a1 + bval[1]);
      r2 = softplus_f(a2 + bval[2]);
    } else {
      r0 = a0 + bvec[o - 3];
      r1 = a1 + bvec[o - 2];
      r2 = a2 + bvec[o - 1];
    }
    vout[t * 13 + o + 0] = r0;
    vout[t * 13 + o + 1] = r1;
    vout[t * 13 + o + 2] = r2;
  }
  __syncthreads();

  // ---- Phase 4: 3x3 MGS-QR + cov = Q L Q^T + eps*I -----------------------
  // Column-sign ambiguity of QR cancels in Q diag(L) Q^T.
  if (tid < TOK_PER_BLK) {
    const float* v = vout + tid * 13;
    float l0 = v[0], l1 = v[1], l2 = v[2];
    float c0x = v[3], c0y = v[6], c0z = v[9];
    float c1x = v[4], c1y = v[7], c1z = v[10];
    float c2x = v[5], c2y = v[8], c2z = v[11];

    float in0 = rsqrtf(fmaxf(c0x*c0x + c0y*c0y + c0z*c0z, 1e-30f));
    float q0x = c0x*in0, q0y = c0y*in0, q0z = c0z*in0;

    float d01 = q0x*c1x + q0y*c1y + q0z*c1z;
    c1x -= d01*q0x; c1y -= d01*q0y; c1z -= d01*q0z;
    float in1 = rsqrtf(fmaxf(c1x*c1x + c1y*c1y + c1z*c1z, 1e-30f));
    float q1x = c1x*in1, q1y = c1y*in1, q1z = c1z*in1;

    float d02 = q0x*c2x + q0y*c2y + q0z*c2z;
    float d12 = q1x*c2x + q1y*c2y + q1z*c2z;
    c2x -= d02*q0x + d12*q1x;
    c2y -= d02*q0y + d12*q1y;
    c2z -= d02*q0z + d12*q1z;
    float in2 = rsqrtf(fmaxf(c2x*c2x + c2y*c2y + c2z*c2z, 1e-30f));
    float q2x = c2x*in2, q2y = c2y*in2, q2z = c2z*in2;

    float m00 = l0*q0x*q0x + l1*q1x*q1x + l2*q2x*q2x + COV_EPS;
    float m11 = l0*q0y*q0y + l1*q1y*q1y + l2*q2y*q2y + COV_EPS;
    float m22 = l0*q0z*q0z + l1*q1z*q1z + l2*q2z*q2z + COV_EPS;
    float m01 = l0*q0x*q0y + l1*q1x*q1y + l2*q2x*q2y;
    float m02 = l0*q0x*q0z + l1*q1x*q1z + l2*q2x*q2z;
    float m12 = l0*q0y*q0z + l1*q1y*q1z + l2*q2y*q2z;

    float* o9 = out + (size_t)(tokBase + tid) * 9;
    o9[0] = m00; o9[1] = m01; o9[2] = m02;
    o9[3] = m01; o9[4] = m11; o9[5] = m12;
    o9[6] = m02; o9[7] = m12; o9[8] = m22;
  }
}

extern "C" void kernel_launch(void* const* d_in, const int* in_sizes, int n_in,
                              void* d_out, int out_size, void* d_ws, size_t ws_size,
                              hipStream_t stream) {
  const float* x    = (const float*)d_in[0];
  const float* W1   = (const float*)d_in[1];
  const float* b1   = (const float*)d_in[2];
  const float* W2   = (const float*)d_in[3];
  const float* b2   = (const float*)d_in[4];
  const float* Wval = (const float*)d_in[5];
  const float* bval = (const float*)d_in[6];
  const float* Wvec = (const float*)d_in[7];
  const float* bvec = (const float*)d_in[8];
  float* out = (float*)d_out;

  __bf16* w1p = (__bf16*)d_ws;              // 512 KB
  __bf16* w2p = w1p + 256 * 1024;           // 512 KB

  repack_w_bf16<<<(256 * 1024) / 256, 256, 0, stream>>>(W1, w1p, 256, 1024);
  repack_w_bf16<<<(1024 * 256) / 256, 256, 0, stream>>>(W2, w2p, 1024, 256);

  const size_t lds_bytes =
      (size_t)TOK_PER_BLK * XA_STRIDE * 2 +   // xa
      (size_t)TOK_PER_BLK * H1_STRIDE * 2 +   // h1
      (size_t)TOK_PER_BLK * HB_STRIDE * 4 +   // xf
      (size_t)TOK_PER_BLK * HB_STRIDE * 4 +   // hb
      (size_t)256 * 12 * 4 +                  // wh
      (size_t)TOK_PER_BLK * 13 * 4;           // vout  (~307 KB < 320 KB WGP)

  (void)hipFuncSetAttribute((const void*)fused_cov_kernel,
                            hipFuncAttributeMaxDynamicSharedMemorySize,
                            (int)lds_bytes);

  const int nblocks = (16 * 4096) / TOK_PER_BLK;  // 1024
  fused_cov_kernel<<<nblocks, 256, lds_bytes, stream>>>(
      x, b1, b2, bval, bvec, Wval, Wvec, w1p, w2p, out);
}
